// GNNEncoder_67740224193371
// MI455X (gfx1250) — matrix operand order repaired and checked by reading, not compiled, not run on hardware
//
#include <hip/hip_runtime.h>
#include <hip/hip_bf16.h>
#include <stdint.h>

#define V_N   512
#define E_NUM 32768
#define F_DIM 8
#define H_DIM 128
#define L_NUM 3
#define G3    384    // 3*H
#define TWO_H 256    // 2*H

typedef __attribute__((ext_vector_type(16))) __bf16 bf16x16;
typedef __attribute__((ext_vector_type(8)))  float  floatx8;

union FragU { bf16x16 v; uint4 q[2]; };

static __device__ __forceinline__ unsigned short f2bf(float f) {
  unsigned int u = __float_as_uint(f);
  u += 0x7FFFu + ((u >> 16) & 1u);       // round-to-nearest-even
  return (unsigned short)(u >> 16);
}

static __device__ __forceinline__ unsigned int pack2bf(float a, float b) {
  return (unsigned int)f2bf(a) | ((unsigned int)f2bf(b) << 16);
}

// Two contiguous 16B loads -> one 16-bit WMMA operand fragment (32 bytes/lane).
static __device__ __forceinline__ bf16x16 frag16(const unsigned short* base, int o0, int o1) {
  FragU t;
  t.q[0] = *reinterpret_cast<const uint4*>(base + o0);
  t.q[1] = *reinterpret_cast<const uint4*>(base + o1);
  return t.v;
}

// ---------------------------------------------------------------- converts
__global__ void k_f32_to_bf16(const float* __restrict__ src,
                              unsigned short* __restrict__ dst, int n) {
  int i = blockIdx.x * blockDim.x + threadIdx.x;
  if (i < n) dst[i] = f2bf(src[i]);
}

// ---------------------------------------------------------------- embedding
// x[ab,h] = sum_f adj[ab,f] * emb[h,f]   (K=8: VALU FMA, memory bound)
__global__ void k_embed(const float* __restrict__ adj, const float* __restrict__ emb,
                        float* __restrict__ x) {
  __shared__ float s_emb[H_DIM * F_DIM];
  __shared__ float s_adj[8 * F_DIM];
  int tid = threadIdx.x;                               // 128 threads
  for (int i = tid; i < H_DIM * F_DIM; i += 128) s_emb[i] = emb[i];
  size_t ab0 = (size_t)blockIdx.x * 8;
  if (tid < 64) s_adj[tid] = adj[ab0 * F_DIM + tid];
  __syncthreads();
  for (int r = 0; r < 8; ++r) {
    float acc = 0.f;
#pragma unroll
    for (int f = 0; f < F_DIM; ++f) acc += s_adj[r * F_DIM + f] * s_emb[tid * F_DIM + f];
    x[(ab0 + r) * H_DIM + tid] = acc;
  }
}

// ---------------------------------------------------------------- col_sum / nnz
__global__ void k_colsum(const float* __restrict__ x, float* __restrict__ col_sum,
                         float* __restrict__ nnz) {
  __shared__ float red[128];
  int b = blockIdx.x, tid = threadIdx.x;               // 128 threads, h = tid
  float s = 0.f, c = 0.f;
  for (int a = 0; a < V_N; ++a) {
    float v = x[((size_t)a * V_N + b) * H_DIM + tid];
    s += v;
    c += (v != 0.f) ? 1.f : 0.f;
  }
  col_sum[b * H_DIM + tid] = s;
  red[tid] = c;
  __syncthreads();
  for (int o = 64; o > 0; o >>= 1) { if (tid < o) red[tid] += red[tid + o]; __syncthreads(); }
  if (tid == 0) nnz[b] = red[0];
}

// ---------------------------------------------------------------- edge messages
// 16 edges per block; msg = relu(x_uv@W1^T + e_in@W2^T + e_out@W3^T) via bf16 WMMA,
// scattered into m at (u,v). Duplicate edges compute identical values -> deterministic.
__global__ void k_message(const float* __restrict__ x, float* __restrict__ m,
                          const float* __restrict__ col_sum, const float* __restrict__ nnz,
                          const long long* __restrict__ edges,
                          const unsigned short* __restrict__ w1b,
                          const unsigned short* __restrict__ w2b,
                          const unsigned short* __restrict__ w3b) {
  __shared__ __align__(16) unsigned short A0[16 * H_DIM];   // x_uv (bf16)
  __shared__ __align__(16) unsigned short A1[16 * H_DIM];   // e_in
  __shared__ __align__(16) unsigned short A2[16 * H_DIM];   // e_out
  __shared__ int   su[16], sv[16];
  __shared__ float cvu[16][16], cuv[16][16];
  __shared__ float rin[16], rout[16];

  int tid = threadIdx.x;                 // 256 threads = 8 waves
  int e   = tid >> 4;                    // edge slot 0..15
  int pp  = tid & 15;
  int e0  = blockIdx.x * 16;

  if (tid < 16) {
    su[tid] = (int)edges[(size_t)(e0 + tid) * 2 + 0];
    sv[tid] = (int)edges[(size_t)(e0 + tid) * 2 + 1];
  }
  __syncthreads();

  int u = su[e], v = sv[e];
  float xu[8], xv[8];
  float c_vu = 0.f, c_uv = 0.f;
  int hbase = pp * 8;
#pragma unroll
  for (int j = 0; j < 8; ++j) {
    int h = hbase + j;
    float a = x[((size_t)u * V_N + v) * H_DIM + h];      // x_uv
    float b = x[((size_t)v * V_N + u) * H_DIM + h];      // x_vu
    xu[j] = a; xv[j] = b;
    c_uv += (a != 0.f) ? 1.f : 0.f;
    c_vu += (b != 0.f) ? 1.f : 0.f;
  }
  cvu[e][pp] = c_vu; cuv[e][pp] = c_uv;
  __syncthreads();
  if (tid < 16) {
    float svu = 0.f, suv = 0.f;
    for (int k = 0; k < 16; ++k) { svu += cvu[tid][k]; suv += cuv[tid][k]; }
    float nin  = (nnz[su[tid]] - svu) * (1.f / H_DIM);
    float nout = (nnz[sv[tid]] - suv) * (1.f / H_DIM);
    if (nin  == 0.f) nin  = 1.f;
    if (nout == 0.f) nout = 1.f;
    rin[tid]  = 1.f / nin;
    rout[tid] = 1.f / nout;
  }
  __syncthreads();
  float ri = rin[e], ro = rout[e];
#pragma unroll
  for (int j = 0; j < 8; ++j) {
    int h = hbase + j;
    float ein  = (col_sum[u * H_DIM + h] - xv[j]) * ri;
    float eout = (col_sum[v * H_DIM + h] - xu[j]) * ro;
    A0[e * H_DIM + h] = f2bf(xu[j]);
    A1[e * H_DIM + h] = f2bf(ein);
    A2[e * H_DIM + h] = f2bf(eout);
  }
  __syncthreads();

  // one 16x16 C tile per wave; accumulate 3 GEMMs x 4 K-chunks
  int wave = tid >> 5, lane = tid & 31;
  int lrow = lane & 15, lhi = lane >> 4;
  int nG   = wave * 16 + lrow;                    // output channel
  floatx8 acc = {};
  const unsigned short* As[3] = { A0, A1, A2 };
  const unsigned short* Ws[3] = { w1b, w2b, w3b };
#pragma unroll
  for (int s = 0; s < 3; ++s) {
    const unsigned short* A = As[s];
    const unsigned short* W = Ws[s];
#pragma unroll
    for (int kc = 0; kc < 4; ++kc) {
      int ka = kc * 32 + lhi * 8;                 // A: lanes>=16 carry K+8 half
      bf16x16 af = frag16(A + lrow * H_DIM, ka, ka + 16);
      int kb = kc * 32 + lhi * 8;                 // B: lanes>=16 carry K+8 half
      bf16x16 bfv = frag16(W + nG * H_DIM, kb, kb + 16);
      acc = __builtin_amdgcn_wmma_f32_16x16x32_bf16(false, af, false, bfv,
                                                    (short)0, acc, false, false);
    }
  }
#pragma unroll
  for (int j = 0; j < 8; ++j) {
    float val = acc[j];
    val = val > 0.f ? val : 0.f;                  // relu
    int er = lhi * 8 + j;                         // edge row of C
    m[((size_t)su[er] * V_N + sv[er]) * H_DIM + nG] = val;
  }
}

// ---------------------------------------------------------------- persistent GRU layer
// One block per 16 batch rows; runs all 512 timesteps. Batch rows are independent,
// so there is no inter-block dependence. Weights (288 KB bf16) + h state live in LDS
// for the whole scan (311 KB of the WGP's 320 KB). Gate accumulators (r/z/n x {xi,gh})
// stay in registers: wave w owns output columns [16w,16w+16) for all three gates,
// which share the same C-tile lane layout. Next-step activation rows are prefetched
// (global_prefetch) during the WMMA phase to hide the strided t-slice gather.
#define GRU_LDS_BYTES (G3*TWO_H*2 + G3*H_DIM*2 + 16*TWO_H*2 + 16*H_DIM*2 + 16*H_DIM*4 + G3*4 + G3*4)

__global__ void __launch_bounds__(256) k_gru_layer(
    float* __restrict__ x, const float* __restrict__ m,
    const unsigned short* __restrict__ wihb, const unsigned short* __restrict__ whhb,
    const float* __restrict__ bih, const float* __restrict__ bhh) {
  extern __shared__ __align__(16) unsigned char smem[];
  unsigned short* sWih = (unsigned short*)smem;            // 384*256 bf16 = 192 KB
  unsigned short* sWhh = sWih + G3 * TWO_H;                // 384*128 bf16 =  96 KB
  unsigned short* sAx  = sWhh + G3 * H_DIM;                // 16*256 bf16  =   8 KB
  unsigned short* sAh  = sAx + 16 * TWO_H;                 // 16*128 bf16  =   4 KB
  float*          sH   = (float*)(sAh + 16 * H_DIM);       // 16*128 f32   =   8 KB
  float*          sBih = sH + 16 * H_DIM;                  // 384 f32
  float*          sBhh = sBih + G3;                        // 384 f32

  int tid = threadIdx.x;                 // 256 threads = 8 waves
  int b0  = blockIdx.x * 16;

  // ---- stage weights + biases once; zero h state
  {
    const uint4* s1 = (const uint4*)wihb; uint4* d1 = (uint4*)sWih;
    for (int i = tid; i < G3 * TWO_H / 8; i += 256) d1[i] = s1[i];
    const uint4* s2 = (const uint4*)whhb; uint4* d2 = (uint4*)sWhh;
    for (int i = tid; i < G3 * H_DIM / 8; i += 256) d2[i] = s2[i];
    for (int i = tid; i < G3; i += 256) { sBih[i] = bih[i]; sBhh[i] = bhh[i]; }
    for (int i = tid; i < 16 * H_DIM; i += 256) sH[i] = 0.f;
  }
  __syncthreads();

  int wave = tid >> 5, lane = tid & 31;
  int lrow = lane & 15, lhi = lane >> 4;
  int cG   = wave * 16 + lrow;           // owned output column 0..127

  for (int t = 0; t < V_N; ++t) {
    // ---- phase 1: stage Ax = bf16([m_t | x_t]) rows, Ah = bf16(h)
    for (int i = tid; i < 16 * 64; i += 256) {           // 16 rows x 64 float4-groups
      int r = i >> 6, c4 = (i & 63) * 4;
      size_t bi = ((size_t)(b0 + r) * V_N + t) * H_DIM;
      const float4 vv = (c4 < H_DIM)
          ? *reinterpret_cast<const float4*>(m + bi + c4)
          : *reinterpret_cast<const float4*>(x + bi + (c4 - H_DIM));
      uint2 o; o.x = pack2bf(vv.x, vv.y); o.y = pack2bf(vv.z, vv.w);
      *reinterpret_cast<uint2*>(sAx + r * TWO_H + c4) = o;
    }
    for (int i = tid; i < 16 * 32; i += 256) {           // 16 rows x 32 float4-groups
      int r = i >> 5, c4 = (i & 31) * 4;
      const float* hp = sH + r * H_DIM + c4;
      uint2 o; o.x = pack2bf(hp[0], hp[1]); o.y = pack2bf(hp[2], hp[3]);
      *reinterpret_cast<uint2*>(sAh + r * H_DIM + c4) = o;
    }
    __syncthreads();

    // ---- prefetch next timestep's m/x rows (128B granularity) while WMMAs run
    if (t + 1 < V_N && tid < 128) {
      int r  = (tid & 63) >> 2;                         // row 0..15
      int ch = (tid & 3) * 32;                          // 128B chunk within 512B row
      size_t bi = ((size_t)(b0 + r) * V_N + (t + 1)) * H_DIM + ch;
      if (tid < 64) __builtin_prefetch(m + bi, 0, 3);
      else          __builtin_prefetch(x + bi, 0, 3);
    }

    // ---- phase 2: six register-resident C tiles per wave
    floatx8 zero = {};
    floatx8 aX[3] = { zero, zero, zero };   // xi:  r, z, n
    floatx8 aH[3] = { zero, zero, zero };   // gh:  r, z, n
#pragma unroll
    for (int p = 0; p < 3; ++p) {
      int nG = p * H_DIM + cG;             // gate row in W (0..383)
#pragma unroll
      for (int kc = 0; kc < 8; ++kc) {     // K = 256 : [m|x] @ Wih^T
        int ka = kc * 32 + lhi * 8;
        bf16x16 af = frag16(sAx + lrow * TWO_H, ka, ka + 16);
        int kb = kc * 32 + lhi * 8;
        bf16x16 bfv = frag16(sWih + nG * TWO_H, kb, kb + 16);
        aX[p] = __builtin_amdgcn_wmma_f32_16x16x32_bf16(false, af, false, bfv,
                                                        (short)0, aX[p], false, false);
      }
#pragma unroll
      for (int kc = 0; kc < 4; ++kc) {     // K = 128 : h @ Whh^T
        int ka = kc * 32 + lhi * 8;
        bf16x16 af = frag16(sAh + lrow * H_DIM, ka, ka + 16);
        int kb = kc * 32 + lhi * 8;
        bf16x16 bfv = frag16(sWhh + nG * H_DIM, kb, kb + 16);
        aH[p] = __builtin_amdgcn_wmma_f32_16x16x32_bf16(false, af, false, bfv,
                                                        (short)0, aH[p], false, false);
      }
    }

    // ---- phase 3: gates entirely in registers (same lane layout for r/z/n tiles)
    float bxr = sBih[cG], bxz = sBih[H_DIM + cG], bxn = sBih[2 * H_DIM + cG];
    float bhr = sBhh[cG], bhz = sBhh[H_DIM + cG], bhn = sBhh[2 * H_DIM + cG];
#pragma unroll
    for (int j = 0; j < 8; ++j) {
      int rloc = lhi * 8 + j;              // local batch row of this C element
      float xr = aX[0][j] + bxr, hr = aH[0][j] + bhr;
      float xz = aX[1][j] + bxz, hz = aH[1][j] + bhz;
      float xn = aX[2][j] + bxn, hn = aH[2][j] + bhn;
      float rg = 1.f / (1.f + __expf(-(xr + hr)));
      float zg = 1.f / (1.f + __expf(-(xz + hz)));
      float ng = tanhf(xn + rg * hn);
      float hp = sH[rloc * H_DIM + cG];
      float hv = (1.f - zg) * ng + zg * hp;
      sH[rloc * H_DIM + cG] = hv;
      x[((size_t)(b0 + rloc) * V_N + t) * H_DIM + cG] = hv;   // ys[b,t,:] in place
    }
    __syncthreads();   // protects sH / sAx / sAh across iterations
  }
}

// ---------------------------------------------------------------- host
extern "C" void kernel_launch(void* const* d_in, const int* in_sizes, int n_in,
                              void* d_out, int out_size, void* d_ws, size_t ws_size,
                              hipStream_t stream) {
  (void)in_sizes; (void)n_in; (void)out_size; (void)ws_size;
  const float*     adj   = (const float*)d_in[0];
  const long long* edges = (const long long*)d_in[1];   // int64 per reference
  const float*     emb   = (const float*)d_in[2];
  const float*     w1    = (const float*)d_in[3];
  const float*     w2    = (const float*)d_in[4];
  const float*     w3    = (const float*)d_in[5];
  const float*     wih   = (const float*)d_in[6];
  const float*     whh   = (const float*)d_in[7];
  const float*     bih   = (const float*)d_in[8];
  const float*     bhh   = (const float*)d_in[9];
  float* x = (float*)d_out;                              // x lives in d_out

  char* ws = (char*)d_ws;
  size_t off = 0;
  float* m        = (float*)(ws + off); off += (size_t)V_N * V_N * H_DIM * 4;  // 128 MB
  float* col_sum  = (float*)(ws + off); off += (size_t)V_N * H_DIM * 4;
  float* nnz      = (float*)(ws + off); off += 4096;
  unsigned short* w1b  = (unsigned short*)(ws + off); off += (size_t)L_NUM * H_DIM * H_DIM * 2;
  unsigned short* w2b  = (unsigned short*)(ws + off); off += (size_t)L_NUM * H_DIM * H_DIM * 2;
  unsigned short* w3b  = (unsigned short*)(ws + off); off += (size_t)L_NUM * H_DIM * H_DIM * 2;
  unsigned short* wihb = (unsigned short*)(ws + off); off += (size_t)G3 * TWO_H * 2;
  unsigned short* whhb = (unsigned short*)(ws + off); off += (size_t)G3 * H_DIM * 2;

  // one-time weight conversion to bf16
  int nmsg = L_NUM * H_DIM * H_DIM;
  k_f32_to_bf16<<<(nmsg + 255) / 256, 256, 0, stream>>>(w1, w1b, nmsg);
  k_f32_to_bf16<<<(nmsg + 255) / 256, 256, 0, stream>>>(w2, w2b, nmsg);
  k_f32_to_bf16<<<(nmsg + 255) / 256, 256, 0, stream>>>(w3, w3b, nmsg);
  int nih = G3 * TWO_H;
  k_f32_to_bf16<<<(nih + 255) / 256, 256, 0, stream>>>(wih, wihb, nih);
  int nhh = G3 * H_DIM;
  k_f32_to_bf16<<<(nhh + 255) / 256, 256, 0, stream>>>(whh, whhb, nhh);

  // embedding
  k_embed<<<V_N * V_N / 8, 128, 0, stream>>>(adj, emb, x);

  for (int l = 0; l < L_NUM; ++l) {
    hipMemcpyAsync(m, x, (size_t)V_N * V_N * H_DIM * 4, hipMemcpyDeviceToDevice, stream);
    k_colsum<<<V_N, 128, 0, stream>>>(x, col_sum, nnz);
    k_message<<<E_NUM / 16, 256, 0, stream>>>(
        x, m, col_sum, nnz, edges,
        w1b + (size_t)l * H_DIM * H_DIM,
        w2b + (size_t)l * H_DIM * H_DIM,
        w3b + (size_t)l * H_DIM * H_DIM);

    // whole 512-step scan in one persistent kernel (batch rows independent)
    k_gru_layer<<<V_N / 16, 256, GRU_LDS_BYTES, stream>>>(x, m, wihb, whhb, bih, bhh);
  }
}
